// GCN_net_678604832929
// MI455X (gfx1250) — matrix operand order repaired and checked by reading, not compiled, not run on hardware
//
#include <hip/hip_runtime.h>
#include <hip/hip_bf16.h>
#include <math.h>

// ---------------------------------------------------------------------------
// Problem constants (match reference)
// ---------------------------------------------------------------------------
#define N_NODES   50000
#define N_EDGES   800000
#define NUM_G     1024
#define IN_DIM    25
#define DIM       64
#define STEPS     3
#define NTILES    3125            // 50000 / 16

// ---------------------------------------------------------------------------
// WMMA types (CDNA5 / gfx1250, wave32)
// ---------------------------------------------------------------------------
typedef __attribute__((ext_vector_type(16))) __bf16 v16bf;
typedef __attribute__((ext_vector_type(8)))  float  v8f;

union ABReg { v16bf v; unsigned int u[8]; };

__device__ __forceinline__ unsigned short f2bf(float f) {
    unsigned int u = __float_as_uint(f);
    u += 0x7FFFu + ((u >> 16) & 1u);       // round-to-nearest-even
    return (unsigned short)(u >> 16);
}
__device__ __forceinline__ unsigned int pack2bf(float lo, float hi) {
    return (unsigned int)f2bf(lo) | ((unsigned int)f2bf(hi) << 16);
}

// monotonic float -> uint key for atomicMax-based segment max
__device__ __forceinline__ unsigned int fkey(float f) {
    unsigned int u = __float_as_uint(f);
    return (u & 0x80000000u) ? ~u : (u | 0x80000000u);
}
__device__ __forceinline__ float unkey(unsigned int k) {
    unsigned int u = (k & 0x80000000u) ? (k ^ 0x80000000u) : ~k;
    return __uint_as_float(u);
}

__device__ __forceinline__ float sigf(float x) { return 1.0f / (1.0f + __expf(-x)); }

// ---------------------------------------------------------------------------
// Generic fill
// ---------------------------------------------------------------------------
__global__ void k_fill(float* p, float v, int n) {
    int i = blockIdx.x * blockDim.x + threadIdx.x;
    if (i < n) p[i] = v;
}

// ---------------------------------------------------------------------------
// Pack W0 (25x64, K-pad to 32) and Wc (64x64) into a contiguous-k B lane order (A uses the same):
//   B 32x16 bf16:  col = lane&15 ; lanes 0-15 hold K=0..15, lanes 16-31 K=16..31
//   VGPR i holds packed pair {K = kb+2i, kb+2i+1},  kb = (lane>>4)*16
// bw0: [nt(4)][lane(32)][i(8)]  uints
// bwc: [kc(2)*4 + nt][lane(32)][i(8)] uints
// ---------------------------------------------------------------------------
__global__ void k_prep_w(const float* __restrict__ W0, const float* __restrict__ Wc,
                         unsigned int* __restrict__ bw0, unsigned int* __restrict__ bwc) {
    int tid = blockIdx.x * blockDim.x + threadIdx.x;
    if (tid < 1024) {                       // W0 pack
        int i = tid & 7, lane = (tid >> 3) & 31, nt = tid >> 8;
        int col = nt * 16 + (lane & 15);
        int k = ((lane >> 4) << 4) + 2 * i;
        float v0 = (k     < IN_DIM) ? W0[k * DIM + col]       : 0.0f;
        float v1 = (k + 1 < IN_DIM) ? W0[(k + 1) * DIM + col] : 0.0f;
        bw0[tid] = pack2bf(v0, v1);
    } else if (tid < 1024 + 2048) {         // Wc pack
        int t = tid - 1024;
        int i = t & 7, lane = (t >> 3) & 31, g = t >> 8;   // g = kc*4 + nt
        int kc = g >> 2, nt = g & 3;
        int col = nt * 16 + (lane & 15);
        int k = kc * 32 + ((lane >> 4) << 4) + 2 * i;
        bwc[t] = pack2bf(Wc[k * DIM + col], Wc[(k + 1) * DIM + col]);
    }
}

// ---------------------------------------------------------------------------
// Fused node GEMM:  hw = relu(x @ W0 + b0) @ Wc
// One wave handles 16 rows. GEMM1: 4 WMMA (K=32 padded). Intermediate staged
// to LDS as bf16, reloaded in A-layout. GEMM2: 4 ntiles x 2 K-chunks = 8 WMMA.
// ---------------------------------------------------------------------------
__global__ __launch_bounds__(128) void k_gemm_nodes(
        const float* __restrict__ x, const float* __restrict__ b0,
        const unsigned int* __restrict__ bw0, const unsigned int* __restrict__ bwc,
        float* __restrict__ hw) {
    __shared__ __align__(16) unsigned short sh[4][16 * DIM];

    const int wave = threadIdx.x >> 5;
    const int lane = threadIdx.x & 31;
    const int tile = blockIdx.x * 4 + wave;
    if (tile >= NTILES) return;                 // wave-uniform exit

    const int lm = lane & 15;                   // row / col within tile
    const int lh = lane >> 4;                   // half select

    // ---- A tile for GEMM1: 16x32 bf16, rows = tile*16 + lm, K padded from 25
    ABReg a;
#pragma unroll
    for (int i = 0; i < 8; ++i) {
        int k = lh * 16 + 2 * i;
        int row = tile * 16 + lm;
        float f0 = (k     < IN_DIM) ? x[row * IN_DIM + k]     : 0.0f;
        float f1 = (k + 1 < IN_DIM) ? x[row * IN_DIM + k + 1] : 0.0f;
        a.u[i] = pack2bf(f0, f1);
    }

    unsigned short* smt = sh[wave];

    // ---- GEMM1 over 4 N-tiles; bias + relu; stage bf16 result into LDS
#pragma unroll
    for (int nt = 0; nt < 4; ++nt) {
        ABReg b;
#pragma unroll
        for (int i = 0; i < 8; ++i) b.u[i] = bw0[(nt * 32 + lane) * 8 + i];
        v8f c = {};
        c = __builtin_amdgcn_wmma_f32_16x16x32_bf16(false, a.v, false, b.v,
                                                    (short)0, c, false, false);
        int col = nt * 16 + lm;
        float bv = b0[col];
#pragma unroll
        for (int rr = 0; rr < 8; ++rr) {
            int m = lh ? (rr + 8) : rr;
            smt[m * DIM + col] = f2bf(fmaxf(c[rr] + bv, 0.0f));
        }
    }

    // fence same-wave LDS store -> load (and block compiler reordering)
    asm volatile("s_wait_dscnt 0" ::: "memory");

    // ---- reload intermediate as two 16x32 A tiles (K chunks 0..31 / 32..63)
    ABReg a0, a1;
#pragma unroll
    for (int i = 0; i < 8; ++i) {
        int k = lh * 16 + 2 * i;     // even -> dword ok
        a0.u[i] = *(const unsigned int*)(smt + lm * DIM + k);
        a1.u[i] = *(const unsigned int*)(smt + lm * DIM + 32 + k);
    }

    // ---- GEMM2: hw tile
#pragma unroll
    for (int nt = 0; nt < 4; ++nt) {
        ABReg bA, bB;
#pragma unroll
        for (int i = 0; i < 8; ++i) {
            bA.u[i] = bwc[((0 * 4 + nt) * 32 + lane) * 8 + i];
            bB.u[i] = bwc[((1 * 4 + nt) * 32 + lane) * 8 + i];
        }
        v8f c = {};
        c = __builtin_amdgcn_wmma_f32_16x16x32_bf16(false, a0.v, false, bA.v,
                                                    (short)0, c, false, false);
        c = __builtin_amdgcn_wmma_f32_16x16x32_bf16(false, a1.v, false, bB.v,
                                                    (short)0, c, false, false);
        int col = nt * 16 + lm;
#pragma unroll
        for (int rr = 0; rr < 8; ++rr) {
            int m = lh ? (rr + 8) : rr;
            hw[(tile * 16 + m) * DIM + col] = c[rr];
        }
    }
}

// ---------------------------------------------------------------------------
// GCN aggregation
// ---------------------------------------------------------------------------
__global__ void k_deg(const int* __restrict__ ei, float* __restrict__ deg, int E) {
    int i = blockIdx.x * blockDim.x + threadIdx.x;
    if (i < E) atomicAdd(&deg[ei[E + i]], 1.0f);      // targets = col
}
__global__ void k_dinv(float* __restrict__ deg, int n) {
    int i = blockIdx.x * blockDim.x + threadIdx.x;
    if (i < n) { float d = deg[i]; deg[i] = (d > 0.0f) ? rsqrtf(d) : 0.0f; }
}
// self-loop term: agg[n] = hw[n] * dinv[n]^2
__global__ void k_agg_init(const float* __restrict__ hw, const float* __restrict__ dinv,
                           float* __restrict__ agg) {
    int gid = blockIdx.x * blockDim.x + threadIdx.x;
    if (gid >= N_NODES * DIM) return;
    int n = gid >> 6;
    float di = dinv[n];
    agg[gid] = hw[gid] * di * di;
}
// edge scatter: 4 threads / edge, 16 dims each
__global__ void k_edge(const int* __restrict__ ei, const float* __restrict__ hw,
                       const float* __restrict__ dinv, float* __restrict__ agg, int E) {
    int gid = blockIdx.x * blockDim.x + threadIdx.x;
    if (gid >= E * 4) return;
    int e = gid >> 2, q = gid & 3;
    int r = ei[e], c = ei[E + e];
    float nrm = dinv[r] * dinv[c];
    const float4* src = (const float4*)(hw + r * DIM + q * 16);
    float* dst = agg + c * DIM + q * 16;
#pragma unroll
    for (int j = 0; j < 4; ++j) {
        float4 v = src[j];
        atomicAdd(dst + j * 4 + 0, v.x * nrm);
        atomicAdd(dst + j * 4 + 1, v.y * nrm);
        atomicAdd(dst + j * 4 + 2, v.z * nrm);
        atomicAdd(dst + j * 4 + 3, v.w * nrm);
    }
}
__global__ void k_bias_relu(float* __restrict__ h, const float* __restrict__ bc) {
    int gid = blockIdx.x * blockDim.x + threadIdx.x;
    if (gid >= N_NODES * DIM) return;
    h[gid] = fmaxf(h[gid] + bc[gid & 63], 0.0f);
}

// ---------------------------------------------------------------------------
// Set2Set LSTM
// ---------------------------------------------------------------------------
__global__ void k_gates(const float* __restrict__ qs, const float* __restrict__ hs,
                        const float* __restrict__ Wih, const float* __restrict__ Whh,
                        const float* __restrict__ bih, const float* __restrict__ bhh,
                        float* __restrict__ gates) {
    int gid = blockIdx.x * blockDim.x + threadIdx.x;
    if (gid >= NUM_G * 4 * DIM) return;
    int b = gid & (NUM_G - 1);
    int j = gid >> 10;
    float acc = bih[j] + bhh[j];
    const float* q  = qs  + b * 2 * DIM;
    const float* wi = Wih + j * 2 * DIM;
#pragma unroll 4
    for (int k = 0; k < 2 * DIM; ++k) acc += q[k] * wi[k];
    const float* hh = hs  + b * DIM;
    const float* wh = Whh + j * DIM;
#pragma unroll 4
    for (int k = 0; k < DIM; ++k) acc += hh[k] * wh[k];
    gates[b * 4 * DIM + j] = acc;
}
__global__ void k_cell(const float* __restrict__ gates, float* __restrict__ hs,
                       float* __restrict__ cs) {
    int gid = blockIdx.x * blockDim.x + threadIdx.x;
    if (gid >= NUM_G * DIM) return;
    int b = gid >> 6, d = gid & 63;
    const float* g = gates + b * 4 * DIM;
    float ig = sigf(g[d]);
    float fg = sigf(g[DIM + d]);
    float gg = tanhf(g[2 * DIM + d]);
    float og = sigf(g[3 * DIM + d]);
    float c = fg * cs[gid] + ig * gg;
    cs[gid] = c;
    hs[gid] = og * tanhf(c);
}

// ---------------------------------------------------------------------------
// Set2Set attention
// ---------------------------------------------------------------------------
__global__ void k_att_e(const float* __restrict__ h, const float* __restrict__ hs,
                        const int* __restrict__ batch, float* __restrict__ e,
                        unsigned int* __restrict__ emaxk) {
    int n = blockIdx.x * blockDim.x + threadIdx.x;
    if (n >= N_NODES) return;
    int b = batch[n];
    const float4* hv = (const float4*)(h + n * DIM);
    const float4* qv = (const float4*)(hs + b * DIM);
    float acc = 0.0f;
#pragma unroll
    for (int j = 0; j < DIM / 4; ++j) {
        float4 a = hv[j], q = qv[j];
        acc += a.x * q.x + a.y * q.y + a.z * q.z + a.w * q.w;
    }
    e[n] = acc;
    atomicMax(&emaxk[b], fkey(acc));
}
__global__ void k_att_a(const int* __restrict__ batch, float* __restrict__ e,
                        const unsigned int* __restrict__ emaxk, float* __restrict__ asum) {
    int n = blockIdx.x * blockDim.x + threadIdx.x;
    if (n >= N_NODES) return;
    int b = batch[n];
    float a = __expf(e[n] - unkey(emaxk[b]));
    e[n] = a;
    atomicAdd(&asum[b], a);
}
__global__ void k_att_r(const int* __restrict__ batch, const float* __restrict__ e,
                        const float* __restrict__ asum, const float* __restrict__ h,
                        float* __restrict__ r) {
    int gid = blockIdx.x * blockDim.x + threadIdx.x;
    if (gid >= N_NODES * 4) return;
    int n = gid >> 2, q = gid & 3;
    int b = batch[n];
    float w = e[n] / asum[b];
    const float4* hv = (const float4*)(h + n * DIM + q * 16);
    float* dst = r + b * DIM + q * 16;
#pragma unroll
    for (int j = 0; j < 4; ++j) {
        float4 v = hv[j];
        atomicAdd(dst + j * 4 + 0, v.x * w);
        atomicAdd(dst + j * 4 + 1, v.y * w);
        atomicAdd(dst + j * 4 + 2, v.z * w);
        atomicAdd(dst + j * 4 + 3, v.w * w);
    }
}
__global__ void k_qstar(const float* __restrict__ hs, const float* __restrict__ r,
                        float* __restrict__ qs) {
    int gid = blockIdx.x * blockDim.x + threadIdx.x;
    if (gid >= NUM_G * DIM) return;
    int b = gid >> 6, d = gid & 63;
    qs[b * 2 * DIM + d]       = hs[gid];
    qs[b * 2 * DIM + DIM + d] = r[gid];
}

// ---------------------------------------------------------------------------
// Output MLP: out = relu(q_star @ W1 + b1) @ W2 + b2
// ---------------------------------------------------------------------------
__global__ void k_out(const float* __restrict__ qs, const float* __restrict__ W1,
                      const float* __restrict__ b1, const float* __restrict__ W2,
                      const float* __restrict__ b2, float* __restrict__ out) {
    int b = blockIdx.x * blockDim.x + threadIdx.x;
    if (b >= NUM_G) return;
    const float* q = qs + b * 2 * DIM;
    float o = 0.0f;
    for (int j = 0; j < DIM; ++j) {
        float a = b1[j];
#pragma unroll 4
        for (int k = 0; k < 2 * DIM; ++k) a += q[k] * W1[k * DIM + j];
        o += fmaxf(a, 0.0f) * W2[j];
    }
    out[b] = o + b2[0];
}

// ---------------------------------------------------------------------------
// Host-side launch orchestration
// ---------------------------------------------------------------------------
static inline int cdiv(long long a, int b) { return (int)((a + b - 1) / b); }

extern "C" void kernel_launch(void* const* d_in, const int* in_sizes, int n_in,
                              void* d_out, int out_size, void* d_ws, size_t ws_size,
                              hipStream_t stream) {
    const float* x    = (const float*)d_in[0];
    const int*   ei   = (const int*)  d_in[1];
    const int*   batch= (const int*)  d_in[2];
    const float* W0   = (const float*)d_in[3];
    const float* b0   = (const float*)d_in[4];
    const float* Wc   = (const float*)d_in[5];
    const float* bc   = (const float*)d_in[6];
    const float* Wih  = (const float*)d_in[7];
    const float* Whh  = (const float*)d_in[8];
    const float* bih  = (const float*)d_in[9];
    const float* bhh  = (const float*)d_in[10];
    const float* W1   = (const float*)d_in[11];
    const float* b1   = (const float*)d_in[12];
    const float* W2   = (const float*)d_in[13];
    const float* b2   = (const float*)d_in[14];
    float* out = (float*)d_out;
    (void)n_in; (void)out_size; (void)ws_size;

    const int E = in_sizes[1] / 2;            // 800000

    // workspace layout (float units)
    float* base = (float*)d_ws;
    float* hw    = base;                                   // 50000*64
    float* h     = hw   + (size_t)N_NODES * DIM;           // agg, then GCN h
    float* dinv  = h    + (size_t)N_NODES * DIM;           // 50000 (deg -> dinv)
    float* e     = dinv + N_NODES;                         // 50000
    float* gates = e    + N_NODES;                         // 1024*256
    float* hs    = gates+ (size_t)NUM_G * 4 * DIM;         // 1024*64
    float* cs    = hs   + (size_t)NUM_G * DIM;
    float* qstar = cs   + (size_t)NUM_G * DIM;             // 1024*128
    float* attblk= qstar+ (size_t)NUM_G * 2 * DIM;         // [emaxk|asum|r]
    unsigned int* emaxk = (unsigned int*)attblk;           // 1024 (key 0 == -inf)
    float* asum  = attblk + NUM_G;                         // 1024
    float* r     = asum   + NUM_G;                         // 1024*64
    unsigned int* bw0 = (unsigned int*)(r + (size_t)NUM_G * DIM);   // 1024
    unsigned int* bwc = bw0 + 1024;                                 // 2048

    const int T = 256;

    // 1) pack weights into WMMA B-layout
    k_prep_w<<<cdiv(3072, T), T, 0, stream>>>(W0, Wc, bw0, bwc);

    // 2) fused node GEMMs with WMMA: hw = relu(x@W0+b0)@Wc
    k_gemm_nodes<<<cdiv(NTILES, 4), 128, 0, stream>>>(x, b0, bw0, bwc, hw);

    // 3) GCN normalization + aggregation
    k_fill<<<cdiv(N_NODES, T), T, 0, stream>>>(dinv, 1.0f, N_NODES);   // self loop
    k_deg <<<cdiv(E, T), T, 0, stream>>>(ei, dinv, E);
    k_dinv<<<cdiv(N_NODES, T), T, 0, stream>>>(dinv, N_NODES);
    k_agg_init<<<cdiv((long long)N_NODES * DIM, T), T, 0, stream>>>(hw, dinv, h);
    k_edge<<<cdiv((long long)E * 4, T), T, 0, stream>>>(ei, hw, dinv, h, E);
    k_bias_relu<<<cdiv((long long)N_NODES * DIM, T), T, 0, stream>>>(h, bc);

    // 4) Set2Set state init (hs, cs, qstar contiguous)
    k_fill<<<cdiv(NUM_G * (DIM + DIM + 2 * DIM), T), T, 0, stream>>>(hs, 0.0f,
            NUM_G * (DIM + DIM + 2 * DIM));

    for (int s = 0; s < STEPS; ++s) {
        k_gates<<<cdiv(NUM_G * 4 * DIM, T), T, 0, stream>>>(qstar, hs, Wih, Whh,
                                                            bih, bhh, gates);
        k_cell <<<cdiv(NUM_G * DIM, T), T, 0, stream>>>(gates, hs, cs);
        // clear [emaxk | asum | r] in one pass (0.0f bits == key 0 == -inf)
        k_fill <<<cdiv(NUM_G * (1 + 1 + DIM), T), T, 0, stream>>>(attblk, 0.0f,
                NUM_G * (1 + 1 + DIM));
        k_att_e<<<cdiv(N_NODES, T), T, 0, stream>>>(h, hs, batch, e, emaxk);
        k_att_a<<<cdiv(N_NODES, T), T, 0, stream>>>(batch, e, emaxk, asum);
        k_att_r<<<cdiv((long long)N_NODES * 4, T), T, 0, stream>>>(batch, e, asum, h, r);
        k_qstar<<<cdiv(NUM_G * DIM, T), T, 0, stream>>>(hs, r, qstar);
    }

    // 5) output MLP
    k_out<<<cdiv(NUM_G, T), T, 0, stream>>>(qstar, W1, b1, W2, b2, out);
}
